// SATLayer_regular_43387759624741
// MI455X (gfx1250) — matrix operand order repaired and checked by reading, not compiled, not run on hardware
//
#include <hip/hip_runtime.h>
#include <hip/hip_bf16.h>

// MI455X / gfx1250: wave32, WMMA 16x16x32 bf16 -> f32 accumulate.
typedef __attribute__((ext_vector_type(16))) __bf16 v16bf;
typedef __attribute__((ext_vector_type(8)))  float  v8f;

#define DIM    256
#define KSTEP  32
#define NKSTEP (DIM / KSTEP)   // 8

// pack two f32 -> two bf16 (truncation; fine for a ReLU'd GEMM feeding a sigmoid gate)
__device__ __forceinline__ unsigned pk2(float a, float b) {
  return (__builtin_bit_cast(unsigned, a) >> 16) |
         (__builtin_bit_cast(unsigned, b) & 0xffff0000u);
}

union BFrag { uint4 u[2]; v16bf v; };

// one 16-byte async global->LDS copy (gfx1250 ASYNCcnt path)
__device__ __forceinline__ void async_cp16(unsigned lds_addr, const void* gaddr) {
  unsigned long long a = (unsigned long long)gaddr;
  asm volatile("global_load_async_to_lds_b128 %0, %1, off"
               :: "v"(lds_addr), "v"(a) : "memory");
}

// f32 -> bf16 pairwise convert (used once for W1..W4; 512 KB total, L2-resident)
__global__ __launch_bounds__(256)
void f32_to_bf16_kernel(const float* __restrict__ src, unsigned* __restrict__ dst, int npairs) {
  const int i = blockIdx.x * 256 + threadIdx.x;
  if (i < npairs) {
    float2 f = ((const float2*)src)[i];
    dst[i] = pk2(f.x, f.y);
  }
}

// Y = relu(X @ W^T + b), fused per-row attention scalar a[row] = Y[row,:].aw + ab.
// Block: 256 thr = 8 waves (4 M x 2 N), tile 64x256; wave tile 16x128 = 8 wmma acc.
// W K-slices (bf16, preconverted) double-buffered in LDS via async-to-LDS copies.
template <bool OUT_F32>
__global__ __launch_bounds__(256)
void gemm_relu_att_kernel(const float* __restrict__ X,
                          const __hip_bfloat16* __restrict__ Wb,  // [256][256] bf16
                          const float* __restrict__ bias,         // [256]
                          const float* __restrict__ aw,           // [256]
                          const float* __restrict__ ab,           // [1]
                          void* __restrict__ Yout,
                          float* __restrict__ aout,               // [nrows]
                          int nrows) {
  __shared__ unsigned short ldsW[2][DIM * KSTEP];  // 2 x 16 KB double buffer
  __shared__ float ldsA[64];                       // per-row attention partials

  const int tid  = threadIdx.x;
  const int lane = tid & 31;
  const int wave = tid >> 5;          // 0..7
  const int wm   = wave >> 1;         // 0..3  (M sub-block of 16 rows)
  const int wn   = wave & 1;          // 0..1  (N half of 128 cols)
  const int half = lane >> 4;         // 0/1
  const int l16  = lane & 15;

  if (tid < 64) ldsA[tid] = 0.f;

  const int rowBase = blockIdx.x * 64 + wm * 16;
  int arow = rowBase + l16;
  if (arow > nrows - 1) arow = nrows - 1;          // clamp; epilogue is guarded
  const float* xrow = X + (size_t)arow * DIM;

  v8f acc[8] = {};

  // issue one K-slice: thread t moves W[t][k0..k0+31] (64B) in 4 async b128 ops
  auto issue_slice = [&](int ks, int buf) {
    const __hip_bfloat16* wsrc = Wb + (size_t)tid * DIM + ks * KSTEP;
    #pragma unroll
    for (int c = 0; c < 4; ++c) {
      unsigned lds = (unsigned)(size_t)&ldsW[buf][tid * KSTEP + c * 8];
      async_cp16(lds, wsrc + c * 8);
    }
  };

  issue_slice(0, 0);

  for (int ks = 0; ks < NKSTEP; ++ks) {
    const int k0 = ks * KSTEP;
    if (ks + 1 < NKSTEP) {
      issue_slice(ks + 1, (ks + 1) & 1);
      asm volatile("s_wait_asynccnt 0x4" ::: "memory");  // slice ks landed (4 in flight ok)
    } else {
      asm volatile("s_wait_asynccnt 0x0" ::: "memory");
    }
    __syncthreads();  // all waves' portions of slice ks visible

    // A fragment (16x32 bf16): half 0 -> K {k0..k0+7, k0+16..k0+23}, half 1 -> +8
    const int c0 = k0 + half * 8;
    float4 f0 = *(const float4*)(xrow + c0);
    float4 f1 = *(const float4*)(xrow + c0 + 4);
    float4 f2 = *(const float4*)(xrow + c0 + 16);
    float4 f3 = *(const float4*)(xrow + c0 + 20);
    BFrag a;
    a.u[0].x = pk2(f0.x, f0.y); a.u[0].y = pk2(f0.z, f0.w);
    a.u[0].z = pk2(f1.x, f1.y); a.u[0].w = pk2(f1.z, f1.w);
    a.u[1].x = pk2(f2.x, f2.y); a.u[1].y = pk2(f2.z, f2.w);
    a.u[1].z = pk2(f3.x, f3.y); a.u[1].w = pk2(f3.z, f3.w);

    const unsigned short* wbuf = ldsW[ks & 1];
    #pragma unroll
    for (int t = 0; t < 8; ++t) {
      // B fragment (32x16): half 0 -> K {k0..k0+7, k0+16..k0+23}, half 1 -> +8
      const int ncol = wn * 128 + t * 16 + l16;
      const uint4* src = (const uint4*)&wbuf[ncol * KSTEP + half * 8];
      BFrag b;
      b.u[0] = src[0];
      b.u[1] = src[2];
      acc[t] = __builtin_amdgcn_wmma_f32_16x16x32_bf16(
          false, a.v, false, b.v, (short)0, acc[t], false, false);
    }
    __syncthreads();  // done reading before this buffer is overwritten
  }

  // Epilogue: bias + ReLU + store + fused attention row-dot.
  // C/D layout: VGPR r, lanes 0-15 -> M=r, lanes 16-31 -> M=r+8; N = l16 (+tile base).
  float part[8];
  #pragma unroll
  for (int r = 0; r < 8; ++r) part[r] = 0.f;

  const bool full = (rowBase + 16) <= nrows;
  if (full) {
    #pragma unroll
    for (int t = 0; t < 8; ++t) {
      const int ncol = wn * 128 + t * 16 + l16;
      const float bn = bias[ncol];
      const float an = aw[ncol];
      const size_t base = (size_t)(rowBase + half * 8) * DIM + ncol;
      #pragma unroll
      for (int r = 0; r < 8; ++r) {
        float v = acc[t][r] + bn;
        v = v > 0.f ? v : 0.f;
        if (OUT_F32) ((float*)Yout)[base + (size_t)r * DIM] = v;
        else         ((__hip_bfloat16*)Yout)[base + (size_t)r * DIM] = __float2bfloat16(v);
        part[r] += v * an;
      }
    }
  } else {
    #pragma unroll
    for (int t = 0; t < 8; ++t) {
      const int ncol = wn * 128 + t * 16 + l16;
      const float bn = bias[ncol];
      const float an = aw[ncol];
      #pragma unroll
      for (int r = 0; r < 8; ++r) {
        const int row = rowBase + r + half * 8;
        if (row < nrows) {
          float v = acc[t][r] + bn;
          v = v > 0.f ? v : 0.f;
          const size_t idx = (size_t)row * DIM + ncol;
          if (OUT_F32) ((float*)Yout)[idx] = v;
          else         ((__hip_bfloat16*)Yout)[idx] = __float2bfloat16(v);
          part[r] += v * an;
        }
      }
    }
  }

  // per-row reduction across lanes/waves via LDS float atomics (ds_add_f32)
  #pragma unroll
  for (int r = 0; r < 8; ++r)
    atomicAdd(&ldsA[wm * 16 + r + half * 8], part[r]);
  __syncthreads();

  if (tid < 64) {
    const int row = blockIdx.x * 64 + tid;
    if (row < nrows) aout[row] = ldsA[tid] + ab[0];
  }
}

// one wave per edge: out[rows[e],:] += sigmoid(ai[r]+aj[c]) * xj[cols[e],:]
// xj is bf16 (three xj matrices together ~154 MB -> resident in 192 MB L2).
__global__ __launch_bounds__(256)
void edge_agg_kernel(const int* __restrict__ rows, const int* __restrict__ cols,
                     const float* __restrict__ ai, const float* __restrict__ aj,
                     const __hip_bfloat16* __restrict__ xj,
                     float* __restrict__ out, int nedges) {
  const int lane = threadIdx.x & 31;
  const int e = blockIdx.x * 8 + (threadIdx.x >> 5);
  if (e >= nedges) return;
  const int r = rows[e];
  const int c = cols[e];
  float att = ai[r] + aj[c];
  att = 1.f / (1.f + __expf(-att));
  const __hip_bfloat16* src = xj + (size_t)c * DIM;
  float* dst = out + (size_t)r * DIM;
  #pragma unroll
  for (int i = 0; i < 8; ++i) {
    const int d = lane + 32 * i;
    unsafeAtomicAdd(dst + d, att * __bfloat162float(src[d]));  // global_atomic_add_f32
  }
}

extern "C" void kernel_launch(void* const* d_in, const int* in_sizes, int n_in,
                              void* d_out, int out_size, void* d_ws, size_t ws_size,
                              hipStream_t stream) {
  const float* x0 = (const float*)d_in[0];
  const float* x1 = (const float*)d_in[1];
  const float* x2 = (const float*)d_in[2];
  const int* rows0 = (const int*)d_in[3];
  const int* cols0 = (const int*)d_in[4];
  const int* rows1 = (const int*)d_in[5];
  const int* cols1 = (const int*)d_in[6];
  const int* rows2 = (const int*)d_in[7];
  const int* cols2 = (const int*)d_in[8];
  const float* W1 = (const float*)d_in[9];  const float* b1 = (const float*)d_in[10];
  const float* W2 = (const float*)d_in[11]; const float* b2 = (const float*)d_in[12];
  const float* W3 = (const float*)d_in[13]; const float* b3 = (const float*)d_in[14];
  const float* W4 = (const float*)d_in[15]; const float* b4 = (const float*)d_in[16];
  const float* a1w = (const float*)d_in[17]; const float* a1b = (const float*)d_in[18];
  const float* a2w = (const float*)d_in[19]; const float* a2b = (const float*)d_in[20];

  const int N0 = in_sizes[0] / DIM;
  const int N1 = in_sizes[1] / DIM;
  const int N2 = in_sizes[2] / DIM;
  const int E0 = in_sizes[3], E1 = in_sizes[5], E2 = in_sizes[7];

  // workspace: bf16 weights (4x128KB), bf16 xj features (~154MB), attention scalars
  char* ws = (char*)d_ws;
  __hip_bfloat16* Wb[4];
  for (int i = 0; i < 4; ++i) { Wb[i] = (__hip_bfloat16*)ws; ws += (size_t)DIM * DIM * 2; }
  __hip_bfloat16* xj0 = (__hip_bfloat16*)ws; ws += (size_t)N0 * DIM * 2;
  __hip_bfloat16* xj1 = (__hip_bfloat16*)ws; ws += (size_t)N1 * DIM * 2;
  __hip_bfloat16* xj2 = (__hip_bfloat16*)ws; ws += (size_t)N2 * DIM * 2;
  float* ai0 = (float*)ws; ws += (size_t)N0 * 4;
  float* aj0 = (float*)ws; ws += (size_t)N0 * 4;
  float* aj1 = (float*)ws; ws += (size_t)N1 * 4;
  float* aj2 = (float*)ws; ws += (size_t)N2 * 4;

  const dim3 blk(256);
  const int wpairs = DIM * DIM / 2;                 // 32768 pairs per weight matrix
  const dim3 wgrid((wpairs + 255) / 256);
  f32_to_bf16_kernel<<<wgrid, blk, 0, stream>>>(W1, (unsigned*)Wb[0], wpairs);
  f32_to_bf16_kernel<<<wgrid, blk, 0, stream>>>(W2, (unsigned*)Wb[1], wpairs);
  f32_to_bf16_kernel<<<wgrid, blk, 0, stream>>>(W3, (unsigned*)Wb[2], wpairs);
  f32_to_bf16_kernel<<<wgrid, blk, 0, stream>>>(W4, (unsigned*)Wb[3], wpairs);

  // xi_0 written fp32 straight into d_out (seeds the segment-sum output)
  gemm_relu_att_kernel<true ><<<dim3((N0 + 63) / 64), blk, 0, stream>>>(x0, Wb[0], b1, a1w, a1b, d_out, ai0, N0);
  gemm_relu_att_kernel<false><<<dim3((N0 + 63) / 64), blk, 0, stream>>>(x0, Wb[1], b2, a2w, a2b, xj0, aj0, N0);
  gemm_relu_att_kernel<false><<<dim3((N1 + 63) / 64), blk, 0, stream>>>(x1, Wb[2], b3, a2w, a2b, xj1, aj1, N1);
  gemm_relu_att_kernel<false><<<dim3((N2 + 63) / 64), blk, 0, stream>>>(x2, Wb[3], b4, a2w, a2b, xj2, aj2, N2);

  edge_agg_kernel<<<dim3((E0 + 7) / 8), blk, 0, stream>>>(rows0, cols0, ai0, aj0, xj0, (float*)d_out, E0);
  edge_agg_kernel<<<dim3((E1 + 7) / 8), blk, 0, stream>>>(rows1, cols1, ai0, aj1, xj1, (float*)d_out, E1);
  edge_agg_kernel<<<dim3((E2 + 7) / 8), blk, 0, stream>>>(rows2, cols2, ai0, aj2, xj2, (float*)d_out, E2);
}
